// IGR_MH_30124900614630
// MI455X (gfx1250) — matrix operand order repaired and checked by reading, not compiled, not run on hardware
//
#include <hip/hip_runtime.h>

// ---------------------------------------------------------------------------
// CDNA5 (gfx1250) WMMA attention block, wave32.
//   D = A(16x32 bf16) x B(32x16 bf16) + C(16x16 f32)
// All GEMM fragments are 16-byte vector loads from pre-converted bf16 layouts;
// no fp32->bf16 conversion and no div/mod inside any K-loop.  Dense GEMMs are
// register-blocked 16(M) x 32(N): one A fragment feeds two WMMAs.
// ---------------------------------------------------------------------------

typedef __attribute__((ext_vector_type(16))) __bf16          v16bf;
typedef __attribute__((ext_vector_type(8)))  float           v8f;
typedef __attribute__((ext_vector_type(8)))  unsigned short  u16x8;

union BF16x16 { v16bf v; u16x8 h[2]; unsigned short u[16]; };

__device__ __forceinline__ unsigned short f2bf(float f) {
    unsigned int x = __float_as_uint(f);
    unsigned int r = (x + 0x7FFFu + ((x >> 16) & 1u)) >> 16;  // RNE
    return (unsigned short)r;
}

// A-matrix (16x32, 16-bit): lane half owns K runs [8h, 8h+8) and [16+8h, 24+8h)
__device__ __forceinline__ void loadA(BF16x16& d, const unsigned short* src, int half) {
    d.h[0] = *(const u16x8*)(src + 8 * half);
    d.h[1] = *(const u16x8*)(src + 16 + 8 * half);
}
// B-matrix (32x16, 16-bit): lane half owns K runs [8h, 8h+8) and [16+8h, 24+8h)
__device__ __forceinline__ void loadB(BF16x16& d, const unsigned short* src, int half) {
    d.h[0] = *(const u16x8*)(src + 8 * half);
    d.h[1] = *(const u16x8*)(src + 16 + 8 * half);
}

__device__ __forceinline__ v8f wmma_bf16(const BF16x16& a, const BF16x16& b, v8f c) {
    return __builtin_amdgcn_wmma_f32_16x16x32_bf16(
        false, a.v, false, b.v, (short)0, c, false, false);
}

// Problem constants
#define BATCH 4
#define CCH   256
#define HH    48
#define WW2   48
#define NTOK  2304          // 48*48
#define NPIX  9216          // BATCH*NTOK
#define HEADS 8
#define DHEAD 32
#define KCONV 2304          // 9*256

// ---------------------------------------------------------------------------
// Layout-conversion kernels (run once per launch; memory-bound, tiny).
// ---------------------------------------------------------------------------
__global__ __launch_bounds__(256) void cvt_bf16(const float* __restrict__ in,
                                                unsigned short* __restrict__ out, int n) {
    int i = blockIdx.x * 256 + threadIdx.x;
    if (i < n) out[i] = f2bf(in[i]);
}

// conv_w (O,C,3,3) -> w_conv_bf[o][tap*256 + c]
__global__ __launch_bounds__(256) void cvt_convw(const float* __restrict__ in,
                                                 unsigned short* __restrict__ out) {
    int i = blockIdx.x * 256 + threadIdx.x;          // 256*2304
    int o = i / KCONV, k = i % KCONV;
    int tap = k >> 8, c = k & 255;
    out[(size_t)o * KCONV + k] = f2bf(in[((size_t)o * CCH + c) * 9 + tap]);
}

// Vx (B,C,H,W) fp32 -> vx_bf (B,H,W,C) bf16
__global__ __launch_bounds__(256) void cvt_vx(const float* __restrict__ in,
                                              unsigned short* __restrict__ out) {
    int i = blockIdx.x * 256 + threadIdx.x;          // NPIX*256
    int c = i & 255, p = i >> 8;                     // p = b*2304 + hw
    int b = p / NTOK, hw = p % NTOK;
    out[(size_t)p * CCH + c] = f2bf(in[((size_t)b * CCH + c) * NTOK + hw]);
}

// ---------------------------------------------------------------------------
// Kernel 1: 3x3 conv (implicit GEMM) + BN + ReLU.
// One wave = 16 out-ch x 32 pixels.  Outer loop: 9 taps (incremental kh/kw,
// per-tap validity + base pointers computed once).  Inner: 8 unrolled
// 32-channel K-steps -> 1 A load + 2 B loads + 2 WMMA each.
// Output: x_t bf16, (B,N,C).
// ---------------------------------------------------------------------------
__global__ __launch_bounds__(256) void conv_bn_relu_wmma(
    const unsigned short* __restrict__ vx_bf, const unsigned short* __restrict__ w_bf,
    const float* __restrict__ conv_b, const float* __restrict__ gamma,
    const float* __restrict__ beta, const float* __restrict__ mean,
    const float* __restrict__ var, unsigned short* __restrict__ x_t)
{
    int wave = (blockIdx.x * blockDim.x + threadIdx.x) >> 5;
    int lane = threadIdx.x & 31;
    int half = lane >> 4, r = lane & 15;
    int mo = wave / 288, ct = wave % 288;
    int o0 = mo * 16;
    int p0 = ct * 32 + r, p1 = p0 + 16;
    int b0 = p0 / NTOK, hw0 = p0 % NTOK, h0 = hw0 / WW2, w0 = hw0 % WW2;
    int b1 = p1 / NTOK, hw1 = p1 % NTOK, h1 = hw1 / WW2, w1 = hw1 % WW2;

    v8f acc0 = {}, acc1 = {};
    const unsigned short* wrow = w_bf + (size_t)(o0 + r) * KCONV;
    const u16x8 zero = {};

    int kh = -1, kw = -1;                     // tap offsets, incremented per tap
    for (int tap = 0; tap < 9; ++tap) {
        int hh0 = h0 + kh, ww0 = w0 + kw;
        int hh1 = h1 + kh, ww1 = w1 + kw;
        bool v0 = (hh0 >= 0) & (hh0 < HH) & (ww0 >= 0) & (ww0 < WW2);
        bool v1 = (hh1 >= 0) & (hh1 < HH) & (ww1 >= 0) & (ww1 < WW2);
        const unsigned short* x0 = vx_bf + (((size_t)b0 * HH + hh0) * WW2 + ww0) * CCH;
        const unsigned short* x1 = vx_bf + (((size_t)b1 * HH + hh1) * WW2 + ww1) * CCH;
        const unsigned short* wt = wrow + tap * CCH;

        #pragma unroll
        for (int cb = 0; cb < 8; ++cb) {
            int c0 = cb << 5;
            BF16x16 A, B0, B1;
            loadA(A, wt + c0, half);
            if (v0) loadB(B0, x0 + c0, half); else { B0.h[0] = zero; B0.h[1] = zero; }
            if (v1) loadB(B1, x1 + c0, half); else { B1.h[0] = zero; B1.h[1] = zero; }
            acc0 = wmma_bf16(A, B0, acc0);
            acc1 = wmma_bf16(A, B1, acc1);
        }
        // advance (kh,kw) without div/mod
        kw = (kw == 1) ? -1 : kw + 1;
        kh = (kw == -1) ? kh + 1 : kh;
    }

    #pragma unroll
    for (int j = 0; j < 8; ++j) {
        int o = o0 + j + 8 * half;
        float inv  = gamma[o] * rsqrtf(var[o] + 1e-5f);
        float addv = (conv_b[o] - mean[o]) * inv + beta[o];
        x_t[(size_t)p0 * CCH + o] = f2bf(fmaxf(acc0[j] * inv + addv, 0.f));
        x_t[(size_t)p1 * CCH + o] = f2bf(fmaxf(acc1[j] * inv + addv, 0.f));
    }
}

// ---------------------------------------------------------------------------
// Kernel 2: QKV GEMM (768 x 256 x 9216) + bias; one wave = 16 o x 32 pixels.
// q,k stored (B,h,N,d); v stored transposed (B,h,d,N).
// ---------------------------------------------------------------------------
__global__ __launch_bounds__(256) void qkv_gemm_wmma(
    const unsigned short* __restrict__ x_t, const unsigned short* __restrict__ qkv_w_bf,
    const float* __restrict__ qkv_b, unsigned short* __restrict__ q_t,
    unsigned short* __restrict__ k_t, unsigned short* __restrict__ v_t)
{
    int wave = (blockIdx.x * blockDim.x + threadIdx.x) >> 5;
    int lane = threadIdx.x & 31, half = lane >> 4, r = lane & 15;
    int mo = wave / 288, ct = wave % 288;
    int o0 = mo * 16;
    int p0 = ct * 32 + r, p1 = p0 + 16;

    v8f acc0 = {}, acc1 = {};
    const unsigned short* wrow  = qkv_w_bf + (size_t)(o0 + r) * CCH;
    const unsigned short* xrow0 = x_t + (size_t)p0 * CCH;
    const unsigned short* xrow1 = x_t + (size_t)p1 * CCH;

    #pragma unroll
    for (int kt = 0; kt < 8; ++kt) {
        int c0 = kt << 5;
        BF16x16 A, B0, B1;
        loadA(A, wrow + c0, half);
        loadB(B0, xrow0 + c0, half);
        loadB(B1, xrow1 + c0, half);
        acc0 = wmma_bf16(A, B0, acc0);
        acc1 = wmma_bf16(A, B1, acc1);
    }

    int b0 = p0 / NTOK, n0 = p0 % NTOK;
    int b1 = p1 / NTOK, n1 = p1 % NTOK;
    #pragma unroll
    for (int j = 0; j < 8; ++j) {
        int o = o0 + j + 8 * half;
        unsigned short bfa = f2bf(acc0[j] + qkv_b[o]);
        unsigned short bfb = f2bf(acc1[j] + qkv_b[o]);
        int head = o / 96, inner = o % 96;
        int which = inner >> 5, dd = inner & 31;
        size_t bh0 = (size_t)b0 * HEADS + head;
        size_t bh1 = (size_t)b1 * HEADS + head;
        if (which == 0) {
            q_t[(bh0 * NTOK + n0) * DHEAD + dd] = bfa;
            q_t[(bh1 * NTOK + n1) * DHEAD + dd] = bfb;
        } else if (which == 1) {
            k_t[(bh0 * NTOK + n0) * DHEAD + dd] = bfa;
            k_t[(bh1 * NTOK + n1) * DHEAD + dd] = bfb;
        } else {
            v_t[(bh0 * DHEAD + dd) * NTOK + n0] = bfa;   // transposed
            v_t[(bh1 * DHEAD + dd) * NTOK + n1] = bfb;
        }
    }
}

// ---------------------------------------------------------------------------
// Kernel 3: flash attention.  One wave = 16 queries of one (b,h); streams
// 32-wide K/V tiles (d=32 == one WMMA K-step), online softmax; P converts
// C-layout -> A-layout through a wave-private LDS patch.
// ---------------------------------------------------------------------------
__global__ __launch_bounds__(256) void attn_wmma(
    const unsigned short* __restrict__ q_t, const unsigned short* __restrict__ k_t,
    const unsigned short* __restrict__ v_t, unsigned short* __restrict__ out_t)
{
    __shared__ unsigned short lds_p[8][16 * 32];   // per-wave P tile (16x32 bf16)

    int wid  = threadIdx.x >> 5;
    int wave = blockIdx.x * 8 + wid;
    int lane = threadIdx.x & 31, half = lane >> 4, r = lane & 15;
    int qrg = wave % 144, bh = wave / 144;
    int n0 = qrg * 16;

    const unsigned short* qb = q_t + (size_t)bh * NTOK * DHEAD;
    const unsigned short* kp = k_t + (size_t)bh * NTOK * DHEAD;
    const unsigned short* vp = v_t + (size_t)bh * DHEAD * NTOK;   // (d, N)
    const float scale = 0.17677669529663687f;                     // 1/sqrt(32)

    BF16x16 Aq;
    loadA(Aq, qb + (size_t)(n0 + r) * DHEAD, half);

    float mrun[8], lrun[8];
    v8f oacc0 = {}, oacc1 = {};
    #pragma unroll
    for (int j = 0; j < 8; ++j) { mrun[j] = -1e30f; lrun[j] = 0.f; }

    unsigned short* pbuf = lds_p[wid];
    const unsigned short* vrow0 = vp + (size_t)r * NTOK;          // e = r
    const unsigned short* vrow1 = vp + (size_t)(16 + r) * NTOK;   // e = 16+r

    for (int mt = 0; mt < 72; ++mt) {
        int m0 = mt << 5;

        // S = Q^T K : two 16x16 tiles
        BF16x16 B0, B1;
        loadB(B0, kp + (size_t)(m0 + r) * DHEAD, half);
        loadB(B1, kp + (size_t)(m0 + 16 + r) * DHEAD, half);
        __builtin_prefetch(kp + (size_t)(m0 + 32 + r) * DHEAD, 0, 3);
        v8f z0 = {}, z1 = {};
        v8f s0 = wmma_bf16(Aq, B0, z0);
        v8f s1 = wmma_bf16(Aq, B1, z1);

        // Online softmax per query row (row j+8*half; 16 cols across lane half)
        #pragma unroll
        for (int j = 0; j < 8; ++j) {
            float a = s0[j] * scale, c = s1[j] * scale;
            float rm = fmaxf(a, c);
            rm = fmaxf(rm, __shfl_xor(rm, 8, 32));
            rm = fmaxf(rm, __shfl_xor(rm, 4, 32));
            rm = fmaxf(rm, __shfl_xor(rm, 2, 32));
            rm = fmaxf(rm, __shfl_xor(rm, 1, 32));
            float mnew = fmaxf(mrun[j], rm);
            float corr = __expf(mrun[j] - mnew);
            float p0 = __expf(a - mnew);
            float p1 = __expf(c - mnew);
            float rs = p0 + p1;
            rs += __shfl_xor(rs, 8, 32);
            rs += __shfl_xor(rs, 4, 32);
            rs += __shfl_xor(rs, 2, 32);
            rs += __shfl_xor(rs, 1, 32);
            lrun[j] = lrun[j] * corr + rs;
            mrun[j] = mnew;
            oacc0[j] *= corr;
            oacc1[j] *= corr;
            int row = j + 8 * half;
            pbuf[row * 32 + r]      = f2bf(p0);
            pbuf[row * 32 + 16 + r] = f2bf(p1);
        }

        // O += P (16x32) * V^T (32 x 32e): two 16-col e tiles
        BF16x16 Ap, Bv0, Bv1;
        loadA(Ap, pbuf + r * 32, half);
        loadB(Bv0, vrow0 + m0, half);
        loadB(Bv1, vrow1 + m0, half);
        oacc0 = wmma_bf16(Ap, Bv0, oacc0);
        oacc1 = wmma_bf16(Ap, Bv1, oacc1);
    }

    int b = bh >> 3, h = bh & 7;
    #pragma unroll
    for (int j = 0; j < 8; ++j) {
        int n = n0 + j + 8 * half;
        float linv = 1.0f / lrun[j];
        size_t base = ((size_t)b * NTOK + n) * CCH + h * DHEAD;
        out_t[base + r]      = f2bf(oacc0[j] * linv);
        out_t[base + 16 + r] = f2bf(oacc1[j] * linv);
    }
}

// ---------------------------------------------------------------------------
// Kernel 4: output projection (256 x 256 x 9216) + bias -> fp32 (B,C,H,W).
// One wave = 16 o x 32 pixels.
// ---------------------------------------------------------------------------
__global__ __launch_bounds__(256) void proj_gemm_wmma(
    const unsigned short* __restrict__ out_t, const unsigned short* __restrict__ proj_w_bf,
    const float* __restrict__ proj_b, float* __restrict__ out)
{
    int wave = (blockIdx.x * blockDim.x + threadIdx.x) >> 5;
    int lane = threadIdx.x & 31, half = lane >> 4, r = lane & 15;
    int mo = wave / 288, ct = wave % 288;
    int o0 = mo * 16;
    int p0 = ct * 32 + r, p1 = p0 + 16;

    v8f acc0 = {}, acc1 = {};
    const unsigned short* wrow  = proj_w_bf + (size_t)(o0 + r) * CCH;
    const unsigned short* xrow0 = out_t + (size_t)p0 * CCH;
    const unsigned short* xrow1 = out_t + (size_t)p1 * CCH;

    #pragma unroll
    for (int kt = 0; kt < 8; ++kt) {
        int c0 = kt << 5;
        BF16x16 A, B0, B1;
        loadA(A, wrow + c0, half);
        loadB(B0, xrow0 + c0, half);
        loadB(B1, xrow1 + c0, half);
        acc0 = wmma_bf16(A, B0, acc0);
        acc1 = wmma_bf16(A, B1, acc1);
    }

    int b0 = p0 / NTOK, n0 = p0 % NTOK;
    int b1 = p1 / NTOK, n1 = p1 % NTOK;
    #pragma unroll
    for (int j = 0; j < 8; ++j) {
        int o = o0 + j + 8 * half;
        float bias = proj_b[o];
        out[((size_t)b0 * CCH + o) * NTOK + n0] = acc0[j] + bias;
        out[((size_t)b1 * CCH + o) * NTOK + n1] = acc1[j] + bias;
    }
}

// ---------------------------------------------------------------------------
extern "C" void kernel_launch(void* const* d_in, const int* in_sizes, int n_in,
                              void* d_out, int out_size, void* d_ws, size_t ws_size,
                              hipStream_t stream) {
    (void)in_sizes; (void)n_in; (void)out_size; (void)ws_size;
    const float* Vx     = (const float*)d_in[0];
    const float* conv_w = (const float*)d_in[1];
    const float* conv_b = (const float*)d_in[2];
    const float* gammap = (const float*)d_in[3];
    const float* betap  = (const float*)d_in[4];
    const float* meanp  = (const float*)d_in[5];
    const float* varp   = (const float*)d_in[6];
    const float* qkv_w  = (const float*)d_in[7];
    const float* qkv_b  = (const float*)d_in[8];
    const float* proj_w = (const float*)d_in[9];
    const float* proj_b = (const float*)d_in[10];

    unsigned char* ws = (unsigned char*)d_ws;
    const size_t SZ = (size_t)NPIX * CCH * sizeof(unsigned short);  // 4,718,592 B
    unsigned short* x_t   = (unsigned short*)(ws + 0 * SZ);  // (B,N,C)
    unsigned short* q_t   = (unsigned short*)(ws + 1 * SZ);  // (B,h,N,d)
    unsigned short* k_t   = (unsigned short*)(ws + 2 * SZ);  // (B,h,N,d)
    unsigned short* v_t   = (unsigned short*)(ws + 3 * SZ);  // (B,h,d,N)
    unsigned short* o_t   = (unsigned short*)(ws + 4 * SZ);  // (B,N,C)
    unsigned short* vx_bf = (unsigned short*)(ws + 5 * SZ);  // (B,H,W,C)
    unsigned short* wconv = (unsigned short*)(ws + 6 * SZ);                      // 256x2304
    unsigned short* wqkv  = (unsigned short*)(ws + 6 * SZ + 1179648);            // 768x256
    unsigned short* wproj = (unsigned short*)(ws + 6 * SZ + 1179648 + 393216);   // 256x256

    // Layout conversions (bf16, WMMA-fragment-friendly)
    cvt_vx   <<<NPIX * CCH / 256, 256, 0, stream>>>(Vx, vx_bf);
    cvt_convw<<<CCH * KCONV / 256, 256, 0, stream>>>(conv_w, wconv);
    cvt_bf16 <<<(3 * CCH * CCH) / 256, 256, 0, stream>>>(qkv_w, wqkv, 3 * CCH * CCH);
    cvt_bf16 <<<(CCH * CCH) / 256, 256, 0, stream>>>(proj_w, wproj, CCH * CCH);

    conv_bn_relu_wmma<<<576, 256, 0, stream>>>(vx_bf, wconv, conv_b, gammap, betap,
                                               meanp, varp, x_t);
    qkv_gemm_wmma<<<1728, 256, 0, stream>>>(x_t, wqkv, qkv_b, q_t, k_t, v_t);
    attn_wmma<<<576, 256, 0, stream>>>(q_t, k_t, v_t, o_t);
    proj_gemm_wmma<<<576, 256, 0, stream>>>(o_t, wproj, proj_b, (float*)d_out);
}
